// GATNet_66640712564982
// MI455X (gfx1250) — matrix operand order repaired and checked by reading, not compiled, not run on hardware
//
#include <hip/hip_runtime.h>
#include <hip/hip_bf16.h>

// ---------------- problem constants (match reference) ----------------
#define N_NODES  20000
#define N_EDGES  320000
#define N_GRAPHS 500
#define IN_DIM   15
#define HC       256          // H*C
#define H_HEADS  4
#define C_DIM    64
#define FC_DIM   256
#define BN_EPS   1e-5f
#define NEG_SLOPE 0.2f

typedef __attribute__((ext_vector_type(16))) __bf16 v16bf;
typedef __attribute__((ext_vector_type(8)))  float  v8f;

// ---------------- helpers ----------------
static __device__ __forceinline__ __bf16 f2bf(float f) {
  // round-to-nearest-even fp32 -> bf16
  unsigned u = __builtin_bit_cast(unsigned, f);
  u += 0x7FFFu + ((u >> 16) & 1u);
  unsigned short h = (unsigned short)(u >> 16);
  return __builtin_bit_cast(__bf16, h);
}

// order-preserving float<->uint key for atomicMax-based segment max
static __device__ __forceinline__ unsigned fkey(float f) {
  unsigned u = __builtin_bit_cast(unsigned, f);
  unsigned mask = (unsigned)((int)u >> 31) | 0x80000000u;
  return u ^ mask;
}
static __device__ __forceinline__ float fdecode(unsigned k) {
  unsigned u = (k & 0x80000000u) ? (k ^ 0x80000000u) : ~k;
  return __builtin_bit_cast(float, u);
}

// ---------------- pad x[N,15] -> xpad[N,32] (zero fill) ----------------
__global__ void pad_x_kernel(const float* __restrict__ x,
                             float* __restrict__ xpad) {
  int idx = blockIdx.x * blockDim.x + threadIdx.x;
  if (idx >= N_NODES * 32) return;
  const int n = idx >> 5, j = idx & 31;
  xpad[idx] = (j < IN_DIM) ? x[n * IN_DIM + j] : 0.0f;
}

// ---------------- pack W[Ksrc,256] (zero-pad to Kpad) into bf16 B-operand layout ---
// Packed layout: Bp[((tn*KSTEPS + t)*32 + lane)*16 + i]
// where per B 32x16 bf16 operand: lane = 16*g + (col&15), elem i<8 -> K = 32t+8g+i, i>=8 -> K = 32t+16+8g+(i-8).
__global__ void packW_kernel(const float* __restrict__ W,
                             __bf16* __restrict__ Bp,
                             int Ksrc, int Kpad) {
  int idx = blockIdx.x * blockDim.x + threadIdx.x;
  if (idx >= Kpad * HC) return;
  const int k = idx >> 8, col = idx & 255;
  const int t = k >> 5, g = (k >> 3) & 1, i = (k & 7) | ((k >> 1) & 8);
  const int tn = col >> 4, ln = (g << 4) | (col & 15);
  const int ksteps = Kpad >> 5;
  const float v = (k < Ksrc) ? W[(size_t)k * HC + col] : 0.0f;
  Bp[(((size_t)tn * ksteps + t) * 32 + ln) * 16 + i] = f2bf(v);
}

// ---------------- WMMA GEMM: Out[20000,256] = A[20000,K] @ W + bias ----------------
// One wave per 16x16 output tile; branch-free, K fully unrolled.
// A loads: four global_load_b128 per K-step; B: one 32B vector load of packed bf16.
template <int KSTEPS>
__global__ void gemm_bias_wmma(const float* __restrict__ A,
                               const __bf16* __restrict__ Bp,
                               const float* __restrict__ bias,
                               float* __restrict__ Out) {
  const int wave = blockIdx.x * (blockDim.x >> 5) + (threadIdx.x >> 5);
  const int lane = threadIdx.x & 31;
  const int tm = wave >> 4;          // 0..1249  (20000/16 exact)
  const int tn = wave & 15;          // 0..15
  const int g  = lane >> 4;          // lane half
  const int mr = lane & 15;
  constexpr int K = KSTEPS * 32;

  const float* Arow = A + (size_t)(tm * 16 + mr) * K;
  const int col = tn * 16 + mr;
  const v16bf* Bt = (const v16bf*)(Bp + (size_t)tn * KSTEPS * 512);

  v8f acc = {};
#pragma unroll
  for (int t = 0; t < KSTEPS; ++t) {
    // A 16x32 bf16 layout: elems 0..7 -> K=32t+8g+i ; elems 8..15 -> K=32t+16+8g+i
    const float4* pa0 = (const float4*)(Arow + t * 32 + 8 * g);
    const float4* pa1 = (const float4*)(Arow + t * 32 + 16 + 8 * g);
    const float4 a0 = pa0[0], a1 = pa0[1];
    const float4 a2 = pa1[0], a3 = pa1[1];
    v16bf a;
    a[0] = f2bf(a0.x); a[1] = f2bf(a0.y); a[2]  = f2bf(a0.z); a[3]  = f2bf(a0.w);
    a[4] = f2bf(a1.x); a[5] = f2bf(a1.y); a[6]  = f2bf(a1.z); a[7]  = f2bf(a1.w);
    a[8] = f2bf(a2.x); a[9] = f2bf(a2.y); a[10] = f2bf(a2.z); a[11] = f2bf(a2.w);
    a[12] = f2bf(a3.x); a[13] = f2bf(a3.y); a[14] = f2bf(a3.z); a[15] = f2bf(a3.w);
    const v16bf b = Bt[t * 32 + lane];
    acc = __builtin_amdgcn_wmma_f32_16x16x32_bf16(
        false, a, false, b, (short)0, acc, false, false);
  }

  const float bv = bias[col];
  // C/D layout: elem r -> row = 16*tm + r + 8g, col = 16*tn + (lane&15)
#pragma unroll
  for (int r = 0; r < 8; ++r)
    Out[(size_t)(tm * 16 + r + 8 * g) * HC + col] = acc[r] + bv;
}

// ---------------- per-(edge,head) attention logits + segment max ----------------
__global__ void edge_logits_kernel(const int* __restrict__ ei,
                                   const float* __restrict__ eattr,
                                   const float* __restrict__ We,   // [3,256]
                                   const float* __restrict__ att,  // [4,64]
                                   const float* __restrict__ xl,
                                   const float* __restrict__ xr,
                                   float* __restrict__ logit,      // [E,4]
                                   unsigned* __restrict__ maxu) {  // [N,4]
  __shared__ float sWe[3 * HC];
  __shared__ float sAtt[HC];
  for (int i = threadIdx.x; i < 3 * HC; i += blockDim.x) sWe[i] = We[i];
  for (int i = threadIdx.x; i < HC; i += blockDim.x) sAtt[i] = att[i];
  __syncthreads();

  int idx = blockIdx.x * blockDim.x + threadIdx.x;
  if (idx >= N_EDGES * H_HEADS) return;
  const int e = idx >> 2, h = idx & 3;
  const int s = ei[e], d = ei[N_EDGES + e];
  const float a0 = eattr[e * 3 + 0], a1 = eattr[e * 3 + 1], a2 = eattr[e * 3 + 2];
  const float* pl = xl + (size_t)s * HC + h * C_DIM;
  const float* pr = xr + (size_t)d * HC + h * C_DIM;
  float acc = 0.0f;
#pragma unroll 4
  for (int c = 0; c < C_DIM; ++c) {
    int j = h * C_DIM + c;
    float et = a0 * sWe[j] + a1 * sWe[HC + j] + a2 * sWe[2 * HC + j];
    float v = pl[c] + pr[c] + et;
    v = (v > 0.0f) ? v : NEG_SLOPE * v;
    acc += v * sAtt[j];
  }
  logit[idx] = acc;
  atomicMax(&maxu[(size_t)d * H_HEADS + h], fkey(acc));
}

// ---------------- exp(logit - segmax) + segment denominator ----------------
__global__ void edge_exp_kernel(const int* __restrict__ ei,
                                const float* __restrict__ logit,
                                const unsigned* __restrict__ maxu,
                                float* __restrict__ exw,
                                float* __restrict__ den) {
  int idx = blockIdx.x * blockDim.x + threadIdx.x;
  if (idx >= N_EDGES * H_HEADS) return;
  const int e = idx >> 2, h = idx & 3;
  const int d = ei[N_EDGES + e];
  const float m = fdecode(maxu[(size_t)d * H_HEADS + h]);
  const float ex = __expf(logit[idx] - m);
  exw[idx] = ex;
  atomicAdd(&den[(size_t)d * H_HEADS + h], ex);
}

// ---------------- weighted scatter-add aggregation ----------------
__global__ void edge_agg_kernel(const int* __restrict__ ei,
                                const float* __restrict__ exw,
                                const float* __restrict__ den,
                                const float* __restrict__ xl,
                                float* __restrict__ agg) {
  long long idx = (long long)blockIdx.x * blockDim.x + threadIdx.x;
  if (idx >= (long long)N_EDGES * HC) return;
  const int e = (int)(idx >> 8);
  const int j = (int)(idx & 255);
  const int h = j >> 6;
  const int s = ei[e], d = ei[N_EDGES + e];
  const float* px = &xl[(size_t)s * HC + j];
  __builtin_prefetch(px, 0, 1);               // global_prefetch_b8
  const float alpha = exw[(size_t)e * H_HEADS + h] /
                      den[(size_t)d * H_HEADS + h];
  atomicAdd(&agg[(size_t)d * HC + j], (*px) * alpha);
}

// ---------------- h = relu(agg + bias), in place ----------------
__global__ void node_finish_kernel(float* __restrict__ hbuf,
                                   const float* __restrict__ bias) {
  int idx = blockIdx.x * blockDim.x + threadIdx.x;
  if (idx >= N_NODES * HC) return;
  const float v = hbuf[idx] + bias[idx & 255];
  hbuf[idx] = v > 0.0f ? v : 0.0f;
}

// ---------------- graph mean pool ----------------
__global__ void pool_sum_kernel(const float* __restrict__ h,
                                const int* __restrict__ batch,
                                float* __restrict__ pool) {
  int idx = blockIdx.x * blockDim.x + threadIdx.x;
  if (idx >= N_NODES * HC) return;
  const int n = idx >> 8, j = idx & 255;
  atomicAdd(&pool[(size_t)batch[n] * HC + j], h[idx]);
}
__global__ void pool_cnt_kernel(const int* __restrict__ batch,
                                float* __restrict__ cnt) {
  int n = blockIdx.x * blockDim.x + threadIdx.x;
  if (n >= N_NODES) return;
  atomicAdd(&cnt[batch[n]], 1.0f);
}
__global__ void pool_div_kernel(float* __restrict__ pool,
                                const float* __restrict__ cnt) {
  int idx = blockIdx.x * blockDim.x + threadIdx.x;
  if (idx >= N_GRAPHS * HC) return;
  const int g = idx >> 8;
  pool[idx] = pool[idx] / fmaxf(cnt[g], 1.0f);
}

// ---------------- fc + eval-mode BN + relu ----------------
__global__ void fc_bn_relu_kernel(const float* __restrict__ in,
                                  const float* __restrict__ W,
                                  const float* __restrict__ b,
                                  const float* __restrict__ gamma,
                                  const float* __restrict__ beta,
                                  const float* __restrict__ mean,
                                  const float* __restrict__ var,
                                  float* __restrict__ out) {
  int idx = blockIdx.x * blockDim.x + threadIdx.x;
  if (idx >= N_GRAPHS * FC_DIM) return;
  const int g = idx >> 8, o = idx & 255;
  float acc = b[o];
  const float* pin = in + (size_t)g * FC_DIM;
#pragma unroll 4
  for (int k = 0; k < FC_DIM; ++k) acc += pin[k] * W[(size_t)k * FC_DIM + o];
  float z = (acc - mean[o]) * rsqrtf(var[o] + BN_EPS) * gamma[o] + beta[o];
  out[idx] = z > 0.0f ? z : 0.0f;
}

__global__ void final_kernel(const float* __restrict__ z,
                             const float* __restrict__ Wlast,  // [256,2]
                             const float* __restrict__ blast,
                             float* __restrict__ out) {
  int idx = blockIdx.x * blockDim.x + threadIdx.x;
  if (idx >= N_GRAPHS * 2) return;
  const int g = idx >> 1, o = idx & 1;
  float acc = blast[o];
  const float* pz = z + (size_t)g * FC_DIM;
#pragma unroll 4
  for (int k = 0; k < FC_DIM; ++k) acc += pz[k] * Wlast[k * 2 + o];
  out[idx] = acc;
}

// ---------------- host orchestration ----------------
static inline size_t align256(size_t v) { return (v + 255) & ~(size_t)255; }

extern "C" void kernel_launch(void* const* d_in, const int* in_sizes, int n_in,
                              void* d_out, int out_size, void* d_ws, size_t ws_size,
                              hipStream_t stream) {
  const float* x     = (const float*)d_in[0];
  const int*   ei    = (const int*)d_in[1];
  const float* eattr = (const float*)d_in[2];
  const int*   batch = (const int*)d_in[3];
  const float* Wl[3]; const float* bl[3]; const float* Wr[3]; const float* br[3];
  const float* We[3]; const float* att[3]; const float* bias[3];
  for (int l = 0; l < 3; ++l) {
    int b0 = 4 + 7 * l;
    Wl[l]   = (const float*)d_in[b0 + 0];
    bl[l]   = (const float*)d_in[b0 + 1];
    Wr[l]   = (const float*)d_in[b0 + 2];
    br[l]   = (const float*)d_in[b0 + 3];
    We[l]   = (const float*)d_in[b0 + 4];
    att[l]  = (const float*)d_in[b0 + 5];
    bias[l] = (const float*)d_in[b0 + 6];
  }
  const float* fc1_W    = (const float*)d_in[25];
  const float* fc1_b    = (const float*)d_in[26];
  const float* fc1_g    = (const float*)d_in[27];
  const float* fc1_beta = (const float*)d_in[28];
  const float* fc1_mean = (const float*)d_in[29];
  const float* fc1_var  = (const float*)d_in[30];
  const float* fc2_W    = (const float*)d_in[31];
  const float* fc2_b    = (const float*)d_in[32];
  const float* fc2_g    = (const float*)d_in[33];
  const float* fc2_beta = (const float*)d_in[34];
  const float* fc2_mean = (const float*)d_in[35];
  const float* fc2_var  = (const float*)d_in[36];
  const float* Wlast    = (const float*)d_in[37];
  const float* blast    = (const float*)d_in[38];
  float* out = (float*)d_out;

  // workspace carve-up (all offsets 256B-aligned)
  char* ws = (char*)d_ws;
  const size_t NB = (size_t)N_NODES * HC * sizeof(float);      // 20.48 MB
  const size_t EB = (size_t)N_EDGES * H_HEADS * sizeof(float); // 5.12 MB
  const size_t MB = (size_t)N_NODES * H_HEADS * sizeof(float); // 320 KB
  const size_t GB = (size_t)N_GRAPHS * HC * sizeof(float);     // 512 KB
  size_t off = 0;
  float*    xl    = (float*)(ws + off); off = align256(off + NB);
  float*    xr    = (float*)(ws + off); off = align256(off + NB);
  float*    hbuf  = (float*)(ws + off); off = align256(off + NB);
  float*    logit = (float*)(ws + off); off = align256(off + EB);
  float*    exw   = (float*)(ws + off); off = align256(off + EB);
  unsigned* maxu  = (unsigned*)(ws + off); off = align256(off + MB);
  float*    den   = (float*)(ws + off); off = align256(off + MB);
  float*    pool  = (float*)(ws + off); off = align256(off + GB);
  float*    cnt   = (float*)(ws + off); off = align256(off + (size_t)N_GRAPHS * sizeof(float));
  float*    z1    = (float*)(ws + off); off = align256(off + GB);
  float*    z2    = (float*)(ws + off); off = align256(off + GB);
  float*    xpad  = (float*)(ws + off); off = align256(off + (size_t)N_NODES * 32 * sizeof(float));
  __bf16*   Bp    = (__bf16*)(ws + off); off = align256(off + (size_t)HC * HC * sizeof(__bf16));
  (void)ws_size; (void)in_sizes; (void)n_in; (void)out_size;

  // launch geometry
  const int gemmBlocks = (20000 /*tiles*/) / 8;               // 8 waves (256 thr)/block
  const int ehBlocks   = (N_EDGES * H_HEADS + 255) / 256;
  const long long aggTotal = (long long)N_EDGES * HC;
  const int aggBlocks  = (int)((aggTotal + 255) / 256);
  const int nodeBlocks = (N_NODES * HC + 255) / 256;

  // zero-pad layer-0 input once: x[N,15] -> xpad[N,32]
  pad_x_kernel<<<(N_NODES * 32 + 255) / 256, 256, 0, stream>>>(x, xpad);

  const float* hin = xpad;
  for (int l = 0; l < 3; ++l) {
    const int Ksrc = (l == 0) ? IN_DIM : HC;
    const int Kpad = (l == 0) ? 32 : HC;
    // xl = hin @ Wl + bl
    packW_kernel<<<(Kpad * HC + 255) / 256, 256, 0, stream>>>(Wl[l], Bp, Ksrc, Kpad);
    if (l == 0) gemm_bias_wmma<1><<<gemmBlocks, 256, 0, stream>>>(hin, Bp, bl[l], xl);
    else        gemm_bias_wmma<8><<<gemmBlocks, 256, 0, stream>>>(hin, Bp, bl[l], xl);
    // xr = hin @ Wr + br
    packW_kernel<<<(Kpad * HC + 255) / 256, 256, 0, stream>>>(Wr[l], Bp, Ksrc, Kpad);
    if (l == 0) gemm_bias_wmma<1><<<gemmBlocks, 256, 0, stream>>>(hin, Bp, br[l], xr);
    else        gemm_bias_wmma<8><<<gemmBlocks, 256, 0, stream>>>(hin, Bp, br[l], xr);

    hipMemsetAsync(maxu, 0, MB, stream);   // key 0 < key(any finite float)
    hipMemsetAsync(den, 0, MB, stream);
    edge_logits_kernel<<<ehBlocks, 256, 0, stream>>>(ei, eattr, We[l], att[l],
                                                     xl, xr, logit, maxu);
    edge_exp_kernel<<<ehBlocks, 256, 0, stream>>>(ei, logit, maxu, exw, den);
    hipMemsetAsync(hbuf, 0, NB, stream);   // layer input no longer needed
    edge_agg_kernel<<<aggBlocks, 256, 0, stream>>>(ei, exw, den, xl, hbuf);
    node_finish_kernel<<<nodeBlocks, 256, 0, stream>>>(hbuf, bias[l]);
    hin = hbuf;
  }

  // global mean pool + FC head
  hipMemsetAsync(pool, 0, GB, stream);
  hipMemsetAsync(cnt, 0, (size_t)N_GRAPHS * sizeof(float), stream);
  pool_sum_kernel<<<nodeBlocks, 256, 0, stream>>>(hbuf, batch, pool);
  pool_cnt_kernel<<<(N_NODES + 255) / 256, 256, 0, stream>>>(batch, cnt);
  pool_div_kernel<<<(N_GRAPHS * HC + 255) / 256, 256, 0, stream>>>(pool, cnt);
  fc_bn_relu_kernel<<<(N_GRAPHS * FC_DIM + 255) / 256, 256, 0, stream>>>(
      pool, fc1_W, fc1_b, fc1_g, fc1_beta, fc1_mean, fc1_var, z1);
  fc_bn_relu_kernel<<<(N_GRAPHS * FC_DIM + 255) / 256, 256, 0, stream>>>(
      z1, fc2_W, fc2_b, fc2_g, fc2_beta, fc2_mean, fc2_var, z2);
  final_kernel<<<(N_GRAPHS * 2 + 255) / 256, 256, 0, stream>>>(z2, Wlast, blast, out);
}
